// TemporalAttention_30391188587395
// MI455X (gfx1250) — compile-verified, matrix operand order checked by reading, not run on hardware
//
#include <hip/hip_runtime.h>

// ---------------------------------------------------------------------------
// MHA layer for MI455X (gfx1250, wave32, WMMA + async global->LDS DMA).
// out  : (8,1024,1024) fp32      at d_out[0 .. 8388608)
// attnW: (8,16,1024,1024) fp32   at d_out[8388608 ..)
// ws   : bf16 Q,K (B,H,T,64), V^T (B,H,64,T), ctx (B,T,1024)  (~67 MB)
//
// Pipeline: qkv_gemm -> fused_attn (scores+softmax+PV strip-resident in LDS,
// P streamed out once with non-temporal stores) -> out_gemm.
// bf16 operand staging uses GLOBAL_LOAD_ASYNC_TO_LDS_B128 (ASYNCcnt).
// ---------------------------------------------------------------------------

typedef __bf16 bf16_t;
typedef __attribute__((ext_vector_type(8)))  bf16_t bf16x8;
typedef __attribute__((ext_vector_type(16))) bf16_t bf16x16;
typedef __attribute__((ext_vector_type(8)))  float  floatx8;
typedef __attribute__((ext_vector_type(4)))  float  floatx4;

#define LDT 40    // LDS leading dim for 32-wide K tiles (80B rows, 16B aligned)
#define QLD 72    // leading dim for 64-wide K tiles (144B rows, 16B aligned)
#define SLD 1032  // S/P strip leading dim (2064B rows, 16B aligned)

static __device__ __forceinline__ floatx8 wmma_bf16(bf16x16 a, bf16x16 b, floatx8 c) {
  // 8 args: (neg_a, A, neg_b, B, c_mod, C, reuse_a, reuse_b)
  return __builtin_amdgcn_wmma_f32_16x16x32_bf16(false, a, false, b, (short)0, c,
                                                 false, false);
}

// Async copy of 16 bytes global -> LDS (VGLOBAL GLOBAL_LOAD_ASYNC_TO_LDS_B128,
// tracked by ASYNCcnt). LDS address = low 32 bits of the generic pointer
// (ISA 10.2: LDS aperture address is addr[31:0]).
static __device__ __forceinline__ void async_cp16(void* lds, const void* g) {
  unsigned int l = (unsigned int)(unsigned long long)lds;
  asm volatile("global_load_async_to_lds_b128 %0, %1, off"
               :: "v"(l), "v"(g) : "memory");
}
static __device__ __forceinline__ void async_wait0() {
  asm volatile("s_wait_asynccnt 0x0" ::: "memory");
}

// Load one 16x32 bf16 operand fragment from an LDS tile stored row-major
// (rows = M for A, rows = N for B-transposed), leading dim ld.
// Register striping per ISA 7.12.2: lanes 0-15 hold K=0..7 & 16..23,
// lanes 16-31 hold K=8..15 & 24..31 (8 contiguous bf16 each -> ds_load_b128).
static __device__ __forceinline__ bf16x16 frag_ld(const bf16_t* base, int ld) {
  const int lane = threadIdx.x & 31;
  const int m    = lane & 15;
  const int kh   = (lane >> 4) << 3;            // 0 or 8
  bf16x8 lo = *reinterpret_cast<const bf16x8*>(base + m * ld + kh);
  bf16x8 hi = *reinterpret_cast<const bf16x8*>(base + m * ld + 16 + kh);
  return __builtin_shufflevector(lo, hi, 0,1,2,3,4,5,6,7,8,9,10,11,12,13,14,15);
}

// ---------------------------------------------------------------------------
// Kernel 1: QKV projection. C(8192x3072) = seq(8192x1024) @ qkv_w + qkv_b,
// scattered to bf16 Q/K in (B,H,T,64) and V in (B,H,64,T) (transposed so the
// attention PV staging is a contiguous async copy). Tile 128x128, K-step 32.
// ---------------------------------------------------------------------------
__global__ __launch_bounds__(256) void qkv_gemm(
    const float* __restrict__ seq, const float* __restrict__ w,
    const float* __restrict__ bias,
    bf16_t* __restrict__ qb, bf16_t* __restrict__ kb, bf16_t* __restrict__ vb) {
  __shared__ __align__(16) bf16_t As[128 * LDT];
  __shared__ __align__(16) bf16_t Bs[128 * LDT];   // stored transposed: [n][k]
  const int tid = threadIdx.x, lane = tid & 31, wid = tid >> 5;
  const int n0 = blockIdx.x * 128, m0 = blockIdx.y * 128;
  const int wr = wid >> 1, wc = wid & 1;
  const int ar = tid >> 1, ac = (tid & 1) * 16;
  const int bk = tid >> 3, bc = (tid & 7) * 16;
  floatx8 acc[2][4] = {};

  for (int k0 = 0; k0 < 1024; k0 += 32) {
    const float4* ga =
        reinterpret_cast<const float4*>(seq + (size_t)(m0 + ar) * 1024 + k0 + ac);
    float4 a0 = ga[0], a1 = ga[1], a2 = ga[2], a3 = ga[3];
    bf16x8 p0 = {(bf16_t)a0.x,(bf16_t)a0.y,(bf16_t)a0.z,(bf16_t)a0.w,
                 (bf16_t)a1.x,(bf16_t)a1.y,(bf16_t)a1.z,(bf16_t)a1.w};
    bf16x8 p1 = {(bf16_t)a2.x,(bf16_t)a2.y,(bf16_t)a2.z,(bf16_t)a2.w,
                 (bf16_t)a3.x,(bf16_t)a3.y,(bf16_t)a3.z,(bf16_t)a3.w};
    *reinterpret_cast<bf16x8*>(&As[ar * LDT + ac])     = p0;
    *reinterpret_cast<bf16x8*>(&As[ar * LDT + ac + 8]) = p1;
    const float4* gb =
        reinterpret_cast<const float4*>(w + (size_t)(k0 + bk) * 3072 + n0 + bc);
    float4 b0 = gb[0], b1 = gb[1], b2 = gb[2], b3 = gb[3];
    float tv[16] = {b0.x,b0.y,b0.z,b0.w, b1.x,b1.y,b1.z,b1.w,
                    b2.x,b2.y,b2.z,b2.w, b3.x,b3.y,b3.z,b3.w};
#pragma unroll
    for (int j = 0; j < 16; ++j) Bs[(bc + j) * LDT + bk] = (bf16_t)tv[j];
    __syncthreads();

    if (k0 + 32 < 1024) {
      __builtin_prefetch(seq + (size_t)(m0 + ar) * 1024 + k0 + 32 + ac, 0, 1);
      __builtin_prefetch(w + (size_t)(k0 + 32 + bk) * 3072 + n0 + bc, 0, 1);
    }

    const bf16_t* ab = &As[(wr * 32) * LDT];
    const bf16_t* bb = &Bs[(wc * 64) * LDT];
    bf16x16 af0 = frag_ld(ab, LDT);
    bf16x16 af1 = frag_ld(ab + 16 * LDT, LDT);
#pragma unroll
    for (int j = 0; j < 4; ++j) {
      bf16x16 bfj = frag_ld(bb + j * 16 * LDT, LDT);
      acc[0][j] = wmma_bf16(af0, bfj, acc[0][j]);
      acc[1][j] = wmma_bf16(af1, bfj, acc[1][j]);
    }
    __syncthreads();
  }

  const int mhi = (lane >> 4) << 3, nl = lane & 15;
#pragma unroll
  for (int fi = 0; fi < 2; ++fi)
#pragma unroll
    for (int fj = 0; fj < 4; ++fj)
#pragma unroll
      for (int r = 0; r < 8; ++r) {
        int gm = m0 + wr * 32 + fi * 16 + mhi + r;
        int gn = n0 + wc * 64 + fj * 16 + nl;
        float val = acc[fi][fj][r] + bias[gn];
        int which = gn >> 10, rem = gn & 1023;
        int h = rem >> 6, d = rem & 63;
        int bIdx = gm >> 10, t = gm & 1023;
        bf16_t bv = (bf16_t)val;
        if (which == 0)
          qb[(((size_t)(bIdx * 16 + h)) * 1024 + t) * 64 + d] = bv;
        else if (which == 1)
          kb[(((size_t)(bIdx * 16 + h)) * 1024 + t) * 64 + d] = bv;
        else  // V stored transposed: (b,h,d,t)
          vb[(((size_t)(bIdx * 16 + h)) * 64 + d) * 1024 + t] = bv;
      }
}

// ---------------------------------------------------------------------------
// Kernel 2 (fused attention): one block per (b,h, 64-query strip).
//  Phase 1: S = Q K^T * 0.125 into LDS strip (bf16), K streamed in 128-key
//           chunks via async global->LDS (K/V stay L2-resident per head).
//  Phase 2: per-wave row softmax in LDS; P written once to d_out with
//           non-temporal stores (537MB stream must not evict L2).
//           Masked query rows -> one-hot at k=0 (reference NaN fallback).
//  Phase 3: O = P V; A-fragments read in place from the LDS strip, V^T
//           chunks async-copied (contiguous rows thanks to the V layout).
// LDS: 132KB strip + 9KB Q + 18KB K/V staging = ~156KB.
// ---------------------------------------------------------------------------
__global__ __launch_bounds__(256) void fused_attn(
    const bf16_t* __restrict__ qbuf, const bf16_t* __restrict__ kbuf,
    const bf16_t* __restrict__ vbuf, const int* __restrict__ mask,
    float* __restrict__ attn, bf16_t* __restrict__ ctx) {
  __shared__ __align__(16) bf16_t Ss[64 * SLD];    // S/P strip, bf16
  __shared__ __align__(16) bf16_t Qs[64 * QLD];    // Q strip 64x64
  __shared__ __align__(16) bf16_t Bs[128 * QLD];   // K chunk / V^T chunk staging
  const int tid = threadIdx.x, lane = tid & 31, wid = tid >> 5;
  const int bh = blockIdx.y, b = bh >> 4, h = bh & 15;
  const int q0 = blockIdx.x * 64;
  const bf16_t* Q = qbuf + (size_t)bh * 1024 * 64;
  const bf16_t* K = kbuf + (size_t)bh * 1024 * 64;
  const bf16_t* V = vbuf + (size_t)bh * 64 * 1024;   // (64, T) row-major

  // stage Q strip 64x64 (async; published by the first phase-1 wait+barrier)
  {
    const int r = tid >> 2, c = (tid & 3) * 16;
    async_cp16(&Qs[r * QLD + c],     Q + (size_t)(q0 + r) * 64 + c);
    async_cp16(&Qs[r * QLD + c + 8], Q + (size_t)(q0 + r) * 64 + c + 8);
  }

  const int wr = wid >> 1, wc = wid & 1;           // wave 4x2 grid for phase 1
  const int mhi = (lane >> 4) << 3, nl = lane & 15;

  // ---- phase 1: scores, 8 chunks of 128 keys
  for (int nc = 0; nc < 8; ++nc) {
    const int n0 = nc * 128;
    __syncthreads();                               // Bs readers of chunk nc-1 done
    {
      const int r = tid >> 1, c = (tid & 1) * 32;
#pragma unroll
      for (int j = 0; j < 4; ++j)
        async_cp16(&Bs[r * QLD + c + j * 8],
                   K + (size_t)(n0 + r) * 64 + c + j * 8);
    }
    async_wait0();
    __syncthreads();
    floatx8 acc[4] = {};
#pragma unroll
    for (int kt = 0; kt < 2; ++kt) {
      bf16x16 af = frag_ld(&Qs[(wr * 16) * QLD + kt * 32], QLD);
#pragma unroll
      for (int j = 0; j < 4; ++j) {
        bf16x16 bfj = frag_ld(&Bs[(wc * 64 + j * 16) * QLD + kt * 32], QLD);
        acc[j] = wmma_bf16(af, bfj, acc[j]);
      }
    }
#pragma unroll
    for (int j = 0; j < 4; ++j)
#pragma unroll
      for (int r = 0; r < 8; ++r) {
        int m = wr * 16 + mhi + r;
        int n = n0 + wc * 64 + j * 16 + nl;
        Ss[m * SLD + n] = (bf16_t)(acc[j][r] * 0.125f);
      }
  }
  __syncthreads();

  // ---- phase 2: softmax, each wave owns 8 rows; 32 cols per lane
#pragma unroll 1
  for (int i = 0; i < 8; ++i) {
    const int qr = wid * 8 + i;
    const int gq = q0 + qr;
    float* prow = attn + ((size_t)bh * 1024 + gq) * 1024;
    if (mask[b * 1024 + gq] == 0) {                // uniform across the wave
      bf16x8 zb = {};
      floatx4 zf = {0.f, 0.f, 0.f, 0.f};
#pragma unroll
      for (int j = 0; j < 4; ++j)
        *reinterpret_cast<bf16x8*>(&Ss[qr * SLD + lane * 32 + j * 8]) = zb;
#pragma unroll
      for (int j = 0; j < 8; ++j)
        __builtin_nontemporal_store(zf, reinterpret_cast<floatx4*>(prow + lane * 32 + j * 4));
      if (lane == 0) {
        Ss[qr * SLD] = (bf16_t)1.0f;
        floatx4 oh = {1.f, 0.f, 0.f, 0.f};
        __builtin_nontemporal_store(oh, reinterpret_cast<floatx4*>(prow));
      }
      continue;
    }
    float x[32];
#pragma unroll
    for (int j = 0; j < 4; ++j) {
      bf16x8 t8 = *reinterpret_cast<const bf16x8*>(&Ss[qr * SLD + lane * 32 + j * 8]);
#pragma unroll
      for (int e = 0; e < 8; ++e) x[j * 8 + e] = (float)t8[e];
    }
    float mx = x[0];
#pragma unroll
    for (int j = 1; j < 32; ++j) mx = fmaxf(mx, x[j]);
#pragma unroll
    for (int o = 16; o > 0; o >>= 1) mx = fmaxf(mx, __shfl_xor(mx, o));
    float sm = 0.f;
#pragma unroll
    for (int j = 0; j < 32; ++j) { x[j] = __expf(x[j] - mx); sm += x[j]; }
#pragma unroll
    for (int o = 16; o > 0; o >>= 1) sm += __shfl_xor(sm, o);
    const float inv = 1.0f / sm;
#pragma unroll
    for (int j = 0; j < 32; ++j) x[j] *= inv;
#pragma unroll
    for (int j = 0; j < 8; ++j) {
      floatx4 f4 = {x[4 * j], x[4 * j + 1], x[4 * j + 2], x[4 * j + 3]};
      __builtin_nontemporal_store(f4, reinterpret_cast<floatx4*>(prow + lane * 32 + j * 4));
    }
#pragma unroll
    for (int j = 0; j < 4; ++j) {
      bf16x8 t8 = {(bf16_t)x[8*j],  (bf16_t)x[8*j+1],(bf16_t)x[8*j+2],(bf16_t)x[8*j+3],
                   (bf16_t)x[8*j+4],(bf16_t)x[8*j+5],(bf16_t)x[8*j+6],(bf16_t)x[8*j+7]};
      *reinterpret_cast<bf16x8*>(&Ss[qr * SLD + lane * 32 + j * 8]) = t8;
    }
  }
  __syncthreads();

  // ---- phase 3: O = P V. Wave w: rows 16*(w>>1), cols 32*(w&1); K-step 32.
  // V^T chunk (64 d-rows x 32 keys) is a contiguous-row async copy.
  const int orow = (wid >> 1) * 16;
  const int ocol = (wid & 1) * 32;
  floatx8 oacc[2] = {};
  for (int k0 = 0; k0 < 1024; k0 += 32) {
    __syncthreads();                               // Bs readers done
    {
      const int vd = tid >> 2, kc = (tid & 3) * 8;
      async_cp16(&Bs[vd * QLD + kc], V + (size_t)vd * 1024 + k0 + kc);
    }
    async_wait0();
    __syncthreads();
    bf16x16 af = frag_ld(&Ss[orow * SLD + k0], SLD);   // P straight from LDS
#pragma unroll
    for (int j = 0; j < 2; ++j) {
      bf16x16 bfj = frag_ld(&Bs[(ocol + j * 16) * QLD], QLD);
      oacc[j] = wmma_bf16(af, bfj, oacc[j]);
    }
  }
#pragma unroll
  for (int j = 0; j < 2; ++j)
#pragma unroll
    for (int r = 0; r < 8; ++r) {
      int gm = q0 + orow + mhi + r;
      int d  = ocol + j * 16 + nl;
      ctx[((size_t)(b * 1024 + gm)) * 1024 + h * 64 + d] = (bf16_t)oacc[j][r];
    }
}

// ---------------------------------------------------------------------------
// Kernel 3: out = ctx(8192x1024 bf16) @ out_w(1024x1024) + out_b -> fp32.
// A-operand staged with async global->LDS (already bf16).
// ---------------------------------------------------------------------------
__global__ __launch_bounds__(256) void out_gemm(
    const bf16_t* __restrict__ ctxb, const float* __restrict__ w,
    const float* __restrict__ bias, float* __restrict__ out) {
  __shared__ __align__(16) bf16_t As[128 * LDT];
  __shared__ __align__(16) bf16_t Bs[128 * LDT];
  const int tid = threadIdx.x, lane = tid & 31, wid = tid >> 5;
  const int n0 = blockIdx.x * 128, m0 = blockIdx.y * 128;
  const int wr = wid >> 1, wc = wid & 1;
  const int ar = tid >> 1, ac = (tid & 1) * 16;
  const int bk = tid >> 3, bc = (tid & 7) * 16;
  floatx8 acc[2][4] = {};

  for (int k0 = 0; k0 < 1024; k0 += 32) {
    // A: async copy (bf16 verbatim); overlaps with the B convert/transpose.
    async_cp16(&As[ar * LDT + ac],
               ctxb + (size_t)(m0 + ar) * 1024 + k0 + ac);
    async_cp16(&As[ar * LDT + ac + 8],
               ctxb + (size_t)(m0 + ar) * 1024 + k0 + ac + 8);
    const float4* gb =
        reinterpret_cast<const float4*>(w + (size_t)(k0 + bk) * 1024 + n0 + bc);
    float4 b0 = gb[0], b1 = gb[1], b2 = gb[2], b3 = gb[3];
    float tv[16] = {b0.x,b0.y,b0.z,b0.w, b1.x,b1.y,b1.z,b1.w,
                    b2.x,b2.y,b2.z,b2.w, b3.x,b3.y,b3.z,b3.w};
#pragma unroll
    for (int j = 0; j < 16; ++j) Bs[(bc + j) * LDT + bk] = (bf16_t)tv[j];
    async_wait0();
    __syncthreads();

    if (k0 + 32 < 1024)
      __builtin_prefetch(w + (size_t)(k0 + 32 + bk) * 1024 + n0 + bc, 0, 1);

    const bf16_t* ab = &As[(wr * 32) * LDT];
    const bf16_t* bb = &Bs[(wc * 64) * LDT];
    bf16x16 af0 = frag_ld(ab, LDT);
    bf16x16 af1 = frag_ld(ab + 16 * LDT, LDT);
#pragma unroll
    for (int j = 0; j < 4; ++j) {
      bf16x16 bfj = frag_ld(bb + j * 16 * LDT, LDT);
      acc[0][j] = wmma_bf16(af0, bfj, acc[0][j]);
      acc[1][j] = wmma_bf16(af1, bfj, acc[1][j]);
    }
    __syncthreads();
  }

  const int mhi = (lane >> 4) << 3, nl = lane & 15;
#pragma unroll
  for (int fi = 0; fi < 2; ++fi)
#pragma unroll
    for (int fj = 0; fj < 4; ++fj)
#pragma unroll
      for (int r = 0; r < 8; ++r) {
        int gm = m0 + wr * 32 + fi * 16 + mhi + r;
        int gn = n0 + wc * 64 + fj * 16 + nl;
        out[(size_t)gm * 1024 + gn] = acc[fi][fj][r] + bias[gn];
      }
}

// ---------------------------------------------------------------------------
extern "C" void kernel_launch(void* const* d_in, const int* in_sizes, int n_in,
                              void* d_out, int out_size, void* d_ws, size_t ws_size,
                              hipStream_t stream) {
  (void)in_sizes; (void)n_in; (void)out_size; (void)ws_size;
  const float* seq    = (const float*)d_in[0];  // (8,1024,1024)
  const int*   mask   = (const int*)  d_in[1];  // (8,1024)
  const float* qkv_w  = (const float*)d_in[2];  // (1024,3072)
  const float* qkv_b  = (const float*)d_in[3];  // (3072,)
  const float* out_w  = (const float*)d_in[4];  // (1024,1024)
  const float* out_b  = (const float*)d_in[5];  // (1024,)

  float* out  = (float*)d_out;
  float* attn = out + (size_t)8 * 1024 * 1024;           // 8388608 floats in

  const size_t QKV_ELEMS = (size_t)8 * 16 * 1024 * 64;   // 8388608 per tensor
  bf16_t* qb  = (bf16_t*)d_ws;
  bf16_t* kb  = qb + QKV_ELEMS;
  bf16_t* vb  = kb + QKV_ELEMS;                          // V^T (B,H,64,T)
  bf16_t* ctx = vb + QKV_ELEMS;                          // (8,1024,1024) bf16

  dim3 blk(256);
  qkv_gemm  <<<dim3(24, 64),  blk, 0, stream>>>(seq, qkv_w, qkv_b, qb, kb, vb);
  fused_attn<<<dim3(16, 128), blk, 0, stream>>>(qb, kb, vb, mask, attn, ctx);
  out_gemm  <<<dim3(8, 64),   blk, 0, stream>>>(ctx, out_w, out_b, out);
}
